// Attention_64665027608793
// MI455X (gfx1250) — matrix operand order repaired and checked by reading, not compiled, not run on hardware
//
#include <hip/hip_runtime.h>

// ---------------------------------------------------------------------------
// CDNA5 WMMA (wave32): v_wmma_f32_16x16x32_bf16
//   A: 16x32 bf16 (v16bf)  lane l: row = l&15, kbase = (l>>4)*8,
//        elem i -> k = kbase + (i&7) + (i>=8 ? 16 : 0)     (two 16B chunks)
//   B: 32x16 bf16 (v16bf)  lane l: col = l&15, elem i -> k = (l>>4)*8 + (i&7) + (i>=8 ? 16 : 0)
//   C/D: 16x16 f32 (v8f)   reg r: lanes 0-15 -> (M=r, N=lane),
//                                 lanes16-31 -> (M=r+8, N=lane-16)
// ---------------------------------------------------------------------------
typedef __bf16 bf16_t;
typedef __attribute__((ext_vector_type(16))) __bf16       v16bf;
typedef __attribute__((ext_vector_type(8)))  float        v8f;
typedef __attribute__((ext_vector_type(4)))  unsigned int v4u;
typedef int v4i_g __attribute__((vector_size(16)));   // matches builtin param

union Frag16 {
    v16bf          v;
    v4u            q[2];
    unsigned short us[16];
};

__device__ __forceinline__ unsigned short f2bf_bits(float f) {
    unsigned u = __builtin_bit_cast(unsigned, f);
    u += 0x7FFFu + ((u >> 16) & 1u);      // round-to-nearest-even
    return (unsigned short)(u >> 16);
}
__device__ __forceinline__ bf16_t f2bf(float f) {
    unsigned short h = f2bf_bits(f);
    return __builtin_bit_cast(bf16_t, h);
}

__device__ __forceinline__ v8f wmma_bf16(v16bf a, v16bf b, v8f c) {
    return __builtin_amdgcn_wmma_f32_16x16x32_bf16(
        false, a, false, b, (short)0, c, false, false);
}

// ---------------------------------------------------------------------------
// CDNA5 sync / async-copy helpers
// ---------------------------------------------------------------------------
__device__ __forceinline__ void block_sync() {        // split workgroup barrier
    asm volatile("s_barrier_signal -1\n\ts_barrier_wait -1" ::: "memory");
}
__device__ __forceinline__ void wait_async0() {
#if __has_builtin(__builtin_amdgcn_s_wait_asynccnt)
    __builtin_amdgcn_s_wait_asynccnt(0);
#else
    asm volatile("s_wait_asynccnt 0" ::: "memory");
#endif
}
__device__ __forceinline__ void lds_fence() {
    asm volatile("s_wait_dscnt 0" ::: "memory");
}

#if __has_builtin(__builtin_amdgcn_global_load_async_to_lds_b128)
#define HAVE_ASYNC_LDS 1
#endif

// 16-byte global -> LDS stage. Async (ASYNCcnt-tracked) when available.
// LDS byte offset = low 32 bits of the generic address (ISA 10.2 aperture rule).
__device__ __forceinline__ void stage_b128(const bf16_t* g, bf16_t* l) {
#ifdef HAVE_ASYNC_LDS
    __builtin_amdgcn_global_load_async_to_lds_b128(
        (__attribute__((address_space(1))) v4i_g*)(unsigned long long)g,
        (__attribute__((address_space(3))) v4i_g*)(unsigned)(unsigned long long)(void*)l,
        0, 0);
#else
    *(v4u*)l = *(const v4u*)g;          // synchronous fallback
#endif
}

// ---------------------------------------------------------------------------
// Constants
// ---------------------------------------------------------------------------
#define BATCH   8
#define SEQ     1024
#define DIM     768
#define NHEADS  12
#define HEADD   64
#define QKVN    (3 * DIM)          // 2304
#define MROWS   (BATCH * SEQ)      // 8192
#define SCALE_F 0.125f             // 64^-0.5

// ---------------------------------------------------------------------------
// Prep kernels (tiny)
// ---------------------------------------------------------------------------
__global__ void __launch_bounds__(256)
cvt_f32_to_bf16(const float* __restrict__ in, bf16_t* __restrict__ out, int n) {
    int i = blockIdx.x * 256 + threadIdx.x;
    if (i < n) out[i] = f2bf(in[i]);
}

// in: [rows][cols] f32 -> out: [cols][rows] bf16 (W^T: B-frags contiguous in K)
__global__ void __launch_bounds__(256)
transpose_f32_to_bf16(const float* __restrict__ in, bf16_t* __restrict__ out,
                      int rows, int cols) {
    int i = blockIdx.x * 256 + threadIdx.x;
    if (i >= rows * cols) return;
    int r = i / cols, c = i % cols;
    out[(size_t)c * rows + r] = f2bf(in[i]);
}

// ---------------------------------------------------------------------------
// Shared WMMA GEMM mainloop: one block = 256 rows x 64 cols, one wave = 32x64.
// W tile (64 cols x 32 K, 4KB) async-staged to LDS, double-buffered, shared by
// all 8 waves. A fragments software-pipelined in registers.
// ---------------------------------------------------------------------------
__device__ __forceinline__ void gemm_mainloop_32x64(
    const bf16_t* __restrict__ arows,   // A base at this wave's row m0, [.,K]
    const bf16_t* __restrict__ wt,      // W^T [N][K]
    int n0, int K,
    bf16_t (*wsh)[64 * 32],             // [2][64*32] LDS double buffer
    v8f (&acc)[2][4])
{
    int tid  = threadIdx.x;
    int lane = tid & 31;
    int lmod = lane & 15, lhi = lane >> 4;

    // staging map: thread t -> W row n0 + t/4, K chunk (t%4)*8  (16B each)
    int srow = tid >> 2, scol = (tid & 3) * 8;
    const bf16_t* wsrc  = wt + (size_t)(n0 + srow) * K + scol;
    bf16_t*       wdst0 = &wsh[0][srow * 32 + scol];
    bf16_t*       wdst1 = &wsh[1][srow * 32 + scol];
    stage_b128(wsrc, wdst0);                       // prologue: k0 = 0

    const bf16_t* arow0 = arows + (size_t)lmod * K + lhi * 8;
    const bf16_t* arow1 = arows + (size_t)(16 + lmod) * K + lhi * 8;
    Frag16 acur[2], anxt[2];
    acur[0].q[0] = *(const v4u*)(arow0);
    acur[0].q[1] = *(const v4u*)(arow0 + 16);
    acur[1].q[0] = *(const v4u*)(arow1);
    acur[1].q[1] = *(const v4u*)(arow1 + 16);

    const int nsteps = K / 32;
    for (int i = 0; i < nsteps; i++) {
        int k0 = i * 32;
        wait_async0();                 // W tile for this step resident
        lds_fence();                   // (covers sync-fallback staging stores)
        block_sync();

        if (i + 1 < nsteps) {          // prefetch next W tile + next A frags
            stage_b128(wsrc + k0 + 32, (i & 1) ? wdst0 : wdst1);
            anxt[0].q[0] = *(const v4u*)(arow0 + k0 + 32);
            anxt[0].q[1] = *(const v4u*)(arow0 + k0 + 48);
            anxt[1].q[0] = *(const v4u*)(arow1 + k0 + 32);
            anxt[1].q[1] = *(const v4u*)(arow1 + k0 + 48);
        }

        const bf16_t* wl = &wsh[i & 1][0];
#pragma unroll
        for (int nb = 0; nb < 4; nb++) {
            const bf16_t* wp = wl + (nb * 16 + lmod) * 32 + lhi * 8;
            Frag16 wb;
            wb.q[0] = *(const v4u*)wp;
            wb.q[1] = *(const v4u*)(wp + 16);
            acc[0][nb] = wmma_bf16(acur[0].v, wb.v, acc[0][nb]);
            acc[1][nb] = wmma_bf16(acur[1].v, wb.v, acc[1][nb]);
        }
        acur[0] = anxt[0];
        acur[1] = anxt[1];
    }
}

// ---------------------------------------------------------------------------
// QKV GEMM: x_bf[8192,768] @ Wt[2304,768] + bias, scattered into
//   q,k : [B,H,N,64] bf16      vt : [B,H,64,N] bf16 (V transposed)
// ---------------------------------------------------------------------------
__global__ void __launch_bounds__(256)
qkv_gemm(const bf16_t* __restrict__ xb, const bf16_t* __restrict__ wt,
         const float* __restrict__ bias,
         bf16_t* __restrict__ qo, bf16_t* __restrict__ ko, bf16_t* __restrict__ vto) {
    __shared__ bf16_t wsh[2][64 * 32];
    int wid = threadIdx.x >> 5, lane = threadIdx.x & 31;
    int lmod = lane & 15, lhi = lane >> 4;
    int blkN = blockIdx.x % (QKVN / 64);
    int blkM = blockIdx.x / (QKVN / 64);
    int n0 = blkN * 64;
    int m0 = blkM * 256 + wid * 32;

    v8f acc[2][4] = {};
    gemm_mainloop_32x64(xb + (size_t)m0 * DIM, wt, n0, DIM, wsh, acc);

    // 64-wide col tile sits in one of {q,k,v} and one head (64 == HEADD).
    int which = n0 / DIM;
    int h     = (n0 % DIM) >> 6;
    int bb    = m0 >> 10;                  // 32 | 1024: one batch per tile
    size_t bh = (size_t)bb * NHEADS + h;
    int rowoff = lhi * 8;
#pragma unroll
    for (int rt = 0; rt < 2; rt++)
#pragma unroll
        for (int nb = 0; nb < 4; nb++) {
            int d = nb * 16 + lmod;
            float bv = bias[n0 + d];
#pragma unroll
            for (int r = 0; r < 8; r++) {
                int n = (m0 + rt * 16 + r + rowoff) & (SEQ - 1);
                float val = acc[rt][nb][r] + bv;
                if (which == 0)      qo[(bh * SEQ + n) * HEADD + d] = f2bf(val);
                else if (which == 1) ko[(bh * SEQ + n) * HEADD + d] = f2bf(val);
                else                 vto[(bh * HEADD + d) * SEQ + n] = f2bf(val);
            }
        }
}

// ---------------------------------------------------------------------------
// Flash attention: block = one (b,h), 8 waves x 32 queries = 256 queries.
// K/V 32-key tiles async-staged to LDS (double-buffered) and shared by all
// waves; online softmax; P goes C-layout -> LDS -> A-layout bf16.
// Output: [B, N, H*64] bf16 (feeds proj GEMM directly).
// ---------------------------------------------------------------------------
__global__ void __launch_bounds__(256)
attn_kernel(const bf16_t* __restrict__ qg, const bf16_t* __restrict__ kg,
            const bf16_t* __restrict__ vtg, bf16_t* __restrict__ og) {
    __shared__ bf16_t kt[2][32 * 64];      // [buf][key(32) x dim(64)]
    __shared__ bf16_t vtl[2][64 * 32];     // [buf][dim(64) x key(32)]
    __shared__ float  pst[8][32 * 32];     // per-wave P staging

    int tid = threadIdx.x;
    int lane = tid & 31, wid = tid >> 5;
    int lmod = lane & 15, lhi = lane >> 4;
    int bh   = blockIdx.x >> 2;
    int qblk = blockIdx.x & 3;
    int h = bh % NHEADS, bb = bh / NHEADS;
    int q0 = qblk * 256 + wid * 32;

    const bf16_t* qb = qg  + (size_t)bh * (SEQ * HEADD);
    const bf16_t* kb = kg  + (size_t)bh * (SEQ * HEADD);
    const bf16_t* vb = vtg + (size_t)bh * (HEADD * SEQ);

    // staging maps: 256 threads x 16B = 4KB per tile
    const bf16_t* ksrc = kb + (size_t)(tid >> 3) * HEADD + (tid & 7) * 8; // +j*64
    const bf16_t* vsrc = vb + (size_t)(tid >> 2) * SEQ + (tid & 3) * 8;   // +j
    bf16_t* kdst[2] = { &kt[0][tid * 8],  &kt[1][tid * 8]  };
    bf16_t* vdst[2] = { &vtl[0][tid * 8], &vtl[1][tid * 8] };
    stage_b128(ksrc, kdst[0]);
    stage_b128(vsrc, vdst[0]);

    // Q A-fragments: [qt][dim-step]
    Frag16 aq[2][2];
#pragma unroll
    for (int qt = 0; qt < 2; qt++) {
        const bf16_t* qp = qb + (size_t)(q0 + qt * 16 + lmod) * HEADD + lhi * 8;
        aq[qt][0].q[0] = *(const v4u*)(qp);
        aq[qt][0].q[1] = *(const v4u*)(qp + 16);
        aq[qt][1].q[0] = *(const v4u*)(qp + 32);
        aq[qt][1].q[1] = *(const v4u*)(qp + 48);
    }

    float m_run[2][8], l_run[2][8];
    v8f o[2][4] = {};
#pragma unroll
    for (int qt = 0; qt < 2; qt++)
#pragma unroll
        for (int r = 0; r < 8; r++) { m_run[qt][r] = -3.0e38f; l_run[qt][r] = 0.0f; }

    float* lp = &pst[wid][0];
    int rowoff = lhi * 8;

    for (int j = 0; j < SEQ; j += 32) {
        int buf = (j >> 5) & 1;
        wait_async0();                 // staged K/V for this step resident
        lds_fence();
        block_sync();
        if (j + 32 < SEQ) {            // prefetch next tiles into other buffer
            stage_b128(ksrc + (size_t)(j + 32) * HEADD, kdst[buf ^ 1]);
            stage_b128(vsrc + (j + 32), vdst[buf ^ 1]);
        }

        // K B-fragments from LDS (shared by both query tiles)
        Frag16 bk[2][2];
#pragma unroll
        for (int t = 0; t < 2; t++)
#pragma unroll
            for (int s = 0; s < 2; s++) {
                const bf16_t* kp = &kt[buf][(t * 16 + lmod) * 64 + s * 32 + lhi * 8];
                bk[t][s].q[0] = *(const v4u*)kp;
                bk[t][s].q[1] = *(const v4u*)(kp + 16);
            }

        float alpha[2][8];
#pragma unroll
        for (int qt = 0; qt < 2; qt++) {
            v8f s0 = {}, s1 = {};
            s0 = wmma_bf16(aq[qt][0].v, bk[0][0].v, s0);
            s0 = wmma_bf16(aq[qt][1].v, bk[0][1].v, s0);
            s1 = wmma_bf16(aq[qt][0].v, bk[1][0].v, s1);
            s1 = wmma_bf16(aq[qt][1].v, bk[1][1].v, s1);

#pragma unroll
            for (int r = 0; r < 8; r++) {
                float sa = s0[r] * SCALE_F, sb = s1[r] * SCALE_F;
                float mx = fmaxf(sa, sb);
                mx = fmaxf(mx, __shfl_xor(mx, 1, 32));
                mx = fmaxf(mx, __shfl_xor(mx, 2, 32));
                mx = fmaxf(mx, __shfl_xor(mx, 4, 32));
                mx = fmaxf(mx, __shfl_xor(mx, 8, 32));
                float mnew = fmaxf(m_run[qt][r], mx);
                alpha[qt][r] = __expf(m_run[qt][r] - mnew);
                float pa = __expf(sa - mnew);
                float pb = __expf(sb - mnew);
                float rs = pa + pb;
                rs += __shfl_xor(rs, 1, 32);
                rs += __shfl_xor(rs, 2, 32);
                rs += __shfl_xor(rs, 4, 32);
                rs += __shfl_xor(rs, 8, 32);
                l_run[qt][r] = l_run[qt][r] * alpha[qt][r] + rs;
                m_run[qt][r] = mnew;
                // stage P (C-layout) into wave-private LDS
                lp[(qt * 16 + r + rowoff) * 32 + lmod]      = pa;
                lp[(qt * 16 + r + rowoff) * 32 + 16 + lmod] = pb;
            }
#pragma unroll
            for (int nb = 0; nb < 4; nb++)
#pragma unroll
                for (int r = 0; r < 8; r++) o[qt][nb][r] *= alpha[qt][r];
        }

        lds_fence();                   // wave-local RAW on P staging
        Frag16 pf[2];
#pragma unroll
        for (int qt = 0; qt < 2; qt++) {
            const float* src = lp + (qt * 16 + lmod) * 32 + lhi * 8;
#pragma unroll
            for (int i = 0; i < 8; i++) pf[qt].us[i]     = f2bf_bits(src[i]);
#pragma unroll
            for (int i = 0; i < 8; i++) pf[qt].us[8 + i] = f2bf_bits(src[16 + i]);
        }

        // O += P @ V  (V B-frag shared by both query tiles)
#pragma unroll
        for (int nb = 0; nb < 4; nb++) {
            const bf16_t* vp = &vtl[buf][(nb * 16 + lmod) * 32 + lhi * 8];
            Frag16 bv;
            bv.q[0] = *(const v4u*)vp;
            bv.q[1] = *(const v4u*)(vp + 16);
            o[0][nb] = wmma_bf16(pf[0].v, bv.v, o[0][nb]);
            o[1][nb] = wmma_bf16(pf[1].v, bv.v, o[1][nb]);
        }
    }

    // normalize + write [B, N, H*64+d] bf16
#pragma unroll
    for (int qt = 0; qt < 2; qt++) {
        float inv[8];
#pragma unroll
        for (int r = 0; r < 8; r++) inv[r] = 1.0f / l_run[qt][r];
#pragma unroll
        for (int nb = 0; nb < 4; nb++)
#pragma unroll
            for (int r = 0; r < 8; r++) {
                int row = q0 + qt * 16 + r + rowoff;
                size_t idx = ((size_t)bb * SEQ + row) * DIM + h * HEADD + nb * 16 + lmod;
                og[idx] = f2bf(o[qt][nb][r] * inv[r]);
            }
    }
}

// ---------------------------------------------------------------------------
// Proj GEMM: attn[8192,768]bf16 @ Wt[768,768]bf16 + bias -> out fp32
// ---------------------------------------------------------------------------
__global__ void __launch_bounds__(256)
proj_gemm(const bf16_t* __restrict__ ab, const bf16_t* __restrict__ wt,
          const float* __restrict__ bias, float* __restrict__ out) {
    __shared__ bf16_t wsh[2][64 * 32];
    int wid = threadIdx.x >> 5, lane = threadIdx.x & 31;
    int lmod = lane & 15, lhi = lane >> 4;
    int blkN = blockIdx.x % (DIM / 64);
    int blkM = blockIdx.x / (DIM / 64);
    int n0 = blkN * 64;
    int m0 = blkM * 256 + wid * 32;

    v8f acc[2][4] = {};
    gemm_mainloop_32x64(ab + (size_t)m0 * DIM, wt, n0, DIM, wsh, acc);

    int rowoff = lhi * 8;
#pragma unroll
    for (int rt = 0; rt < 2; rt++)
#pragma unroll
        for (int nb = 0; nb < 4; nb++) {
            float bv = bias[n0 + nb * 16 + lmod];
#pragma unroll
            for (int r = 0; r < 8; r++) {
                size_t idx = (size_t)(m0 + rt * 16 + r + rowoff) * DIM + n0 + nb * 16 + lmod;
                out[idx] = acc[rt][nb][r] + bv;
            }
        }
}

// ---------------------------------------------------------------------------
// Launch
// ---------------------------------------------------------------------------
extern "C" void kernel_launch(void* const* d_in, const int* in_sizes, int n_in,
                              void* d_out, int out_size, void* d_ws, size_t ws_size,
                              hipStream_t stream) {
    (void)in_sizes; (void)n_in; (void)out_size; (void)ws_size;

    const float* x      = (const float*)d_in[0];
    const float* w_qkv  = (const float*)d_in[1];
    const float* b_qkv  = (const float*)d_in[2];
    const float* w_proj = (const float*)d_in[3];
    const float* b_proj = (const float*)d_in[4];
    float*       out    = (float*)d_out;

    char* ws = (char*)d_ws;
    const size_t SZ_XBF  = (size_t)MROWS * DIM * 2;
    const size_t SZ_WQKV = (size_t)QKVN * DIM * 2;
    const size_t SZ_WPRJ = (size_t)DIM * DIM * 2;
    const size_t SZ_HEAD = (size_t)BATCH * NHEADS * SEQ * HEADD * 2;

    bf16_t* x_bf    = (bf16_t*)(ws);
    bf16_t* wt_qkv  = (bf16_t*)(ws + SZ_XBF);
    bf16_t* wt_proj = (bf16_t*)(ws + SZ_XBF + SZ_WQKV);
    bf16_t* qbuf    = (bf16_t*)(ws + SZ_XBF + SZ_WQKV + SZ_WPRJ);
    bf16_t* kbuf    = (bf16_t*)(ws + SZ_XBF + SZ_WQKV + SZ_WPRJ + SZ_HEAD);
    bf16_t* vtbuf   = (bf16_t*)(ws + SZ_XBF + SZ_WQKV + SZ_WPRJ + 2 * SZ_HEAD);
    bf16_t* attnbuf = (bf16_t*)(ws + SZ_XBF + SZ_WQKV + SZ_WPRJ + 3 * SZ_HEAD);

    {
        int n = MROWS * DIM;
        cvt_f32_to_bf16<<<(n + 255) / 256, 256, 0, stream>>>(x, x_bf, n);
    }
    {
        int n = DIM * QKVN;
        transpose_f32_to_bf16<<<(n + 255) / 256, 256, 0, stream>>>(w_qkv, wt_qkv, DIM, QKVN);
    }
    {
        int n = DIM * DIM;
        transpose_f32_to_bf16<<<(n + 255) / 256, 256, 0, stream>>>(w_proj, wt_proj, DIM, DIM);
    }

    // QKV projection: block = 256 rows x 64 cols
    qkv_gemm<<<(MROWS / 256) * (QKVN / 64), 256, 0, stream>>>(
        x_bf, wt_qkv, b_qkv, qbuf, kbuf, vtbuf);

    // Flash attention: block = one (b,h) x 256 queries
    attn_kernel<<<(BATCH * NHEADS) * (SEQ / 256), 256, 0, stream>>>(
        qbuf, kbuf, vtbuf, attnbuf);

    // Output projection
    proj_gemm<<<(MROWS / 256) * (DIM / 64), 256, 0, stream>>>(
        attnbuf, wt_proj, b_proj, out);
}
